// PointTxFormer_223338299441
// MI455X (gfx1250) — compile-verified, matrix operand order checked by reading, not run on hardware
//
#include <hip/hip_runtime.h>
#include <hip/hip_bf16.h>

// ---------------- problem constants (match reference) ----------------
#define NN 50000
#define MM 600000
#define GG 2000
// E = 128, DIN = 64, DEI = 16, DT = 64, DC = 64, HID = 128, DOUT = 3, NB = 2

typedef __attribute__((ext_vector_type(16))) __bf16 v16bf;
typedef __attribute__((ext_vector_type(8)))  float  v8f;

// ---------------- helpers ----------------
__device__ __forceinline__ float fastrcp(float x) { return __builtin_amdgcn_rcpf(x); }
__device__ __forceinline__ float siluf(float x) { return x * fastrcp(1.f + __expf(-x)); }

__device__ __forceinline__ unsigned short f2bf(float x) {
  unsigned u = __float_as_uint(x);
  unsigned r = u + 0x7FFFu + ((u >> 16) & 1u);   // round-to-nearest-even
  return (unsigned short)(r >> 16);
}
__device__ __forceinline__ float bf2f(unsigned short h) {
  return __uint_as_float(((unsigned)h) << 16);
}
// order-preserving f32 <-> u32 for atomicMax
__device__ __forceinline__ unsigned f32_ord(float x) {
  unsigned b = __float_as_uint(x);
  return (b & 0x80000000u) ? ~b : (b | 0x80000000u);
}
__device__ __forceinline__ float ord_f32(unsigned k) {
  unsigned b = (k & 0x80000000u) ? (k & 0x7FFFFFFFu) : ~k;
  return __uint_as_float(b);
}

__device__ __forceinline__ v8f wmma_bf16(v16bf a, v16bf b, v8f c) {
  // (neg_a, A, neg_b, B, c_mod, C, reuse_a, reuse_b)
  return __builtin_amdgcn_wmma_f32_16x16x32_bf16(false, a, false, b, (short)0, c, false, false);
}

// A fragment: 16x32 bf16 tile, row-major, row stride ld (elements).
// ISA layout: lane m=lane%16, half=lane/16; dword j holds K pair:
//   j 0..3 -> K = 2j + 8*half ; j 4..7 -> K = 16 + 2(j-4) + 8*half
__device__ __forceinline__ v16bf frag_a(const unsigned short* __restrict__ A,
                                        int ld, int kOff, int lane) {
  int m = lane & 15, half = (lane >> 4) & 1;
  union { v16bf v; unsigned u[8]; } f;
  const unsigned short* p = A + m * ld + kOff + half * 8;
#pragma unroll
  for (int j = 0; j < 8; ++j) {
    int k = ((j & 4) << 2) + ((j & 3) << 1);
    f.u[j] = (unsigned)p[k] | ((unsigned)p[k + 1] << 16);
  }
  return f.v;
}

// B fragment from PRE-SWIZZLED weights: dwords for (kc, tile, lane) are
// contiguous -> two b128 loads. Layout: dst[(((kc*ctiles)+tile)*32+lane)*8 + j]
__device__ __forceinline__ v16bf frag_bp(const unsigned* __restrict__ Bp,
                                         int kc, int tile, int ctiles, int lane) {
  union { v16bf v; unsigned u[8]; } f;
  const unsigned* p = Bp + ((((long long)kc * ctiles + tile) * 32 + lane) << 3);
#pragma unroll
  for (int j = 0; j < 8; ++j) f.u[j] = p[j];
  return f.v;
}

// ---------------- conversion / packing kernels ----------------
__global__ void k_cvt_pad(const float* __restrict__ src, unsigned short* __restrict__ dst,
                          int r_in, int c_in, int r_out, int c_out) {
  long long i = blockIdx.x * 256LL + threadIdx.x;
  long long tot = (long long)r_out * c_out;
  if (i >= tot) return;
  int r = (int)(i / c_out), c = (int)(i % c_out);
  dst[i] = (r < r_in && c < c_in) ? f2bf(src[(long long)r * c_in + c]) : (unsigned short)0;
}

// pack an [r_in x c_in] f32 weight (used as WMMA B) into fragment-order bf16
__global__ void k_pack_b(const float* __restrict__ src, unsigned* __restrict__ dst,
                         int r_in, int c_in, int kchunks, int ctiles) {
  long long i = blockIdx.x * 256LL + threadIdx.x;
  long long tot = (long long)kchunks * ctiles * 256;   // 32 lanes * 8 dwords
  if (i >= tot) return;
  int j    = (int)(i & 7);
  int lane = (int)((i >> 3) & 31);
  int tile = (int)((i >> 8) % ctiles);
  int kc   = (int)((i >> 8) / ctiles);
  int kl = ((j & 4) << 2) + ((j & 3) << 1) + ((lane >> 4) & 1) * 8;
  int kg = kc * 32 + kl;
  int n  = tile * 16 + (lane & 15);
  unsigned short lo = (kg     < r_in && n < c_in) ? f2bf(src[(long long)kg       * c_in + n]) : (unsigned short)0;
  unsigned short hi = (kg + 1 < r_in && n < c_in) ? f2bf(src[(long long)(kg + 1) * c_in + n]) : (unsigned short)0;
  dst[i] = (unsigned)lo | ((unsigned)hi << 16);
}

__global__ void k_tembed(const float* __restrict__ t, const float* __restrict__ W_t,
                         float* __restrict__ tembed) {
  int n = threadIdx.x;                     // 128 threads
  float s = 0.f;
#pragma unroll 8
  for (int k = 0; k < 64; ++k) s += t[k] * W_t[k * 128 + n];
  tembed[n] = siluf(s);
}

// pack u = [virtual | c7 | t] -> bf16 [G,256]
__global__ void k_upack(const float* __restrict__ virt, const float* __restrict__ c7,
                        const float* __restrict__ t, unsigned short* __restrict__ u_bf) {
  int i = blockIdx.x * 256 + threadIdx.x;  // G*256 total
  int g = i >> 8, c = i & 255;
  float v;
  if (c < 128)      v = virt[(long long)g * 128 + c];
  else if (c < 192) v = c7[(long long)g * 64 + (c - 128)];
  else              v = t[c - 192];
  u_bf[i] = f2bf(v);
}

__global__ void k_deg(const int* __restrict__ ei, float* __restrict__ deg) {
  int i = blockIdx.x * 256 + threadIdx.x;
  if (i < MM) atomicAdd(&deg[ei[MM + i]], 1.f);
}

// ---------------- h = silu(nodes@W_nin) + tembed ----------------
__global__ void k_embed_nodes(const unsigned short* __restrict__ nodes_bf,
                              const unsigned* __restrict__ Wnin_p,
                              const float* __restrict__ tembed,
                              float* __restrict__ h, unsigned short* __restrict__ h_bf) {
  __shared__ unsigned short At[16 * 64];
  int r0 = blockIdx.x * 16, tid = threadIdx.x;
  for (int i = tid; i < 16 * 64; i += 256) At[i] = nodes_bf[(long long)r0 * 64 + i];
  __syncthreads();
  int wave = tid >> 5, lane = tid & 31, n0 = wave * 16;
  v8f acc = {};
#pragma unroll
  for (int kc = 0; kc < 2; ++kc)
    acc = wmma_bf16(frag_a(At, 64, kc * 32, lane), frag_bp(Wnin_p, kc, wave, 8, lane), acc);
  int n = n0 + (lane & 15), half = (lane >> 4) & 1;
#pragma unroll
  for (int r = 0; r < 8; ++r) {
    long long gi = (long long)(r0 + r + half * 8) * 128 + n;
    float v = siluf(acc[r]) + tembed[n];
    h[gi] = v; h_bf[gi] = f2bf(v);
  }
}

// ---------------- e = silu(edges@W_ein) (K padded 16->32) ----------------
__global__ void k_embed_edges(const unsigned short* __restrict__ edges_bf, // [M,32] padded
                              const unsigned* __restrict__ Wein_p,         // packed [32,128]
                              unsigned short* __restrict__ e_bf) {
  int e0 = blockIdx.x * 16;
  int wave = threadIdx.x >> 5, lane = threadIdx.x & 31, n0 = wave * 16;
  const unsigned short* A = edges_bf + (long long)e0 * 32;
  v8f acc = {};
  acc = wmma_bf16(frag_a(A, 32, 0, lane), frag_bp(Wein_p, 0, wave, 8, lane), acc);
  int n = n0 + (lane & 15), half = (lane >> 4) & 1;
#pragma unroll
  for (int r = 0; r < 8; ++r)
    e_bf[(long long)(e0 + r + half * 8) * 128 + n] = f2bf(siluf(acc[r]));
}

// ---------------- q,k,v = h @ {Wq,Wk,Wv} ----------------
__global__ void k_qkv(const unsigned short* __restrict__ h_bf,
                      const unsigned* __restrict__ Wq_p,
                      const unsigned* __restrict__ Wk_p,
                      const unsigned* __restrict__ Wv_p,
                      unsigned short* __restrict__ q_bf,
                      unsigned short* __restrict__ k_bf,
                      unsigned short* __restrict__ v_bf) {
  __shared__ unsigned short At[16 * 128];
  int r0 = blockIdx.x * 16, tid = threadIdx.x;
  for (int i = tid; i < 2048; i += 256) At[i] = h_bf[(long long)r0 * 128 + i];
  __syncthreads();
  int wave = tid >> 5, lane = tid & 31, n0 = wave * 16;
  v16bf a[4];
#pragma unroll
  for (int kc = 0; kc < 4; ++kc) a[kc] = frag_a(At, 128, kc * 32, lane);
  v8f accq = {}, acck = {}, accv = {};
#pragma unroll
  for (int kc = 0; kc < 4; ++kc) {
    accq = wmma_bf16(a[kc], frag_bp(Wq_p, kc, wave, 8, lane), accq);
    acck = wmma_bf16(a[kc], frag_bp(Wk_p, kc, wave, 8, lane), acck);
    accv = wmma_bf16(a[kc], frag_bp(Wv_p, kc, wave, 8, lane), accv);
  }
  int n = n0 + (lane & 15), half = (lane >> 4) & 1;
#pragma unroll
  for (int r = 0; r < 8; ++r) {
    long long gi = (long long)(r0 + r + half * 8) * 128 + n;
    q_bf[gi] = f2bf(accq[r]);
    k_bf[gi] = f2bf(acck[r]);
    v_bf[gi] = f2bf(accv[r]);
  }
}

// ---------------- virt_out = virtual @ Wvirt ----------------
__global__ void k_virt(const unsigned short* __restrict__ virt_bf,
                       const unsigned* __restrict__ Wv_p,
                       float* __restrict__ virt_out) {
  int r0 = blockIdx.x * 16;
  int wave = threadIdx.x >> 5, lane = threadIdx.x & 31, n0 = wave * 16;
  const unsigned short* A = virt_bf + (long long)r0 * 128;
  v8f acc = {};
#pragma unroll
  for (int kc = 0; kc < 4; ++kc)
    acc = wmma_bf16(frag_a(A, 128, kc * 32, lane), frag_bp(Wv_p, kc, wave, 8, lane), acc);
  int n = n0 + (lane & 15), half = (lane >> 4) & 1;
#pragma unroll
  for (int r = 0; r < 8; ++r)
    virt_out[(long long)(r0 + r + half * 8) * 128 + n] = acc[r];
}

// ---------- pass 1: logits=(q[col]-k[row]+delta+e)@Wg, val=v[row]+delta+e, seg-max ----------
__global__ void k_edge_pass1(const int* __restrict__ ei, const float* __restrict__ x,
                             const unsigned short* __restrict__ q_bf,
                             const unsigned short* __restrict__ k_bf,
                             const unsigned short* __restrict__ v_bf,
                             const unsigned short* __restrict__ e_bf,
                             const float* __restrict__ Wpos, const float* __restrict__ bpos,
                             const unsigned* __restrict__ Wg_p,
                             unsigned short* __restrict__ logits_bf,
                             unsigned short* __restrict__ val_bf,
                             unsigned* __restrict__ nodemax) {
  __shared__ int srow[16], scol[16];
  __shared__ unsigned short pre[16 * 128];
  int e0 = blockIdx.x * 16, tid = threadIdx.x;
  if (tid < 16) { srow[tid] = ei[e0 + tid]; scol[tid] = ei[MM + e0 + tid]; }
  __syncthreads();
  for (int i = tid; i < 2048; i += 256) {
    int m = i >> 7, n = i & 127;
    int row = srow[m], col = scol[m];
    float rx = x[row * 3 + 0] - x[col * 3 + 0];
    float ry = x[row * 3 + 1] - x[col * 3 + 1];
    float rz = x[row * 3 + 2] - x[col * 3 + 2];
    float delta = siluf(rx * Wpos[n] + ry * Wpos[128 + n] + rz * Wpos[256 + n] + bpos[n]);
    long long eidx = (long long)(e0 + m) * 128 + n;
    float ev = bf2f(e_bf[eidx]);
    float p = bf2f(q_bf[(long long)col * 128 + n]) - bf2f(k_bf[(long long)row * 128 + n]) + delta + ev;
    float vv = bf2f(v_bf[(long long)row * 128 + n]) + delta + ev;
    pre[i] = f2bf(p);
    val_bf[eidx] = f2bf(vv);
  }
  __syncthreads();
  int wave = tid >> 5, lane = tid & 31, n0 = wave * 16;
  v8f acc = {};
#pragma unroll
  for (int kc = 0; kc < 4; ++kc)
    acc = wmma_bf16(frag_a(pre, 128, kc * 32, lane), frag_bp(Wg_p, kc, wave, 8, lane), acc);
  int n = n0 + (lane & 15), half = (lane >> 4) & 1;
#pragma unroll
  for (int r = 0; r < 8; ++r) {
    int m = r + half * 8;
    long long eidx = (long long)(e0 + m) * 128 + n;
    logits_bf[eidx] = f2bf(acc[r]);
    atomicMax(&nodemax[(long long)scol[m] * 128 + n], f32_ord(acc[r]));
  }
}

// ---------- pass 2: ex = exp(logit - max); accumulate ex and ex*val ----------
__global__ void k_edge_pass2(const int* __restrict__ ei,
                             const unsigned short* __restrict__ logits_bf,
                             const unsigned short* __restrict__ val_bf,
                             const unsigned* __restrict__ nodemax,
                             float* __restrict__ expsum, float* __restrict__ agg) {
  long long stride = (long long)gridDim.x * blockDim.x;
  long long tot = (long long)MM * 128;
  for (long long idx = blockIdx.x * (long long)blockDim.x + threadIdx.x; idx < tot; idx += stride) {
    int i = (int)(idx >> 7), n = (int)(idx & 127);
    int col = ei[MM + i];
    float mx = ord_f32(nodemax[(long long)col * 128 + n]);
    if (!(mx == mx) || fabsf(mx) > 1e37f) mx = 0.f;   // where(isfinite(m), m, 0)
    float ex = __expf(bf2f(logits_bf[idx]) - mx);
    atomicAdd(&expsum[(long long)col * 128 + n], ex);
    atomicAdd(&agg[(long long)col * 128 + n], ex * bf2f(val_bf[idx]));
  }
}

// ---------- h += silu((agg/expsum)@Wo) + virt_out[batch] ----------
__global__ void k_node_update(const float* __restrict__ agg, const float* __restrict__ expsum,
                              const unsigned* __restrict__ Wo_p,
                              const float* __restrict__ virt_out, const int* __restrict__ nbatch,
                              float* __restrict__ h, unsigned short* __restrict__ h_bf) {
  __shared__ unsigned short At[16 * 128];
  int r0 = blockIdx.x * 16, tid = threadIdx.x;
  for (int i = tid; i < 2048; i += 256) {
    long long gi = (long long)r0 * 128 + i;
    At[i] = f2bf(agg[gi] * fastrcp(expsum[gi] + 1e-9f));
  }
  __syncthreads();
  int wave = tid >> 5, lane = tid & 31, n0 = wave * 16;
  v8f acc = {};
#pragma unroll
  for (int kc = 0; kc < 4; ++kc)
    acc = wmma_bf16(frag_a(At, 128, kc * 32, lane), frag_bp(Wo_p, kc, wave, 8, lane), acc);
  int n = n0 + (lane & 15), half = (lane >> 4) & 1;
#pragma unroll
  for (int r = 0; r < 8; ++r) {
    int row = r0 + r + half * 8;
    long long gi = (long long)row * 128 + n;
    float v = h[gi] + siluf(acc[r]) + virt_out[(long long)nbatch[row] * 128 + n];
    h[gi] = v; h_bf[gi] = f2bf(v);
  }
}

// ---------- e += silu([h[row]|h[col]|e]@Weu); last block also builds agg_e ----------
__global__ void k_edge_update(const int* __restrict__ ei,
                              const unsigned short* __restrict__ h_bf,
                              const unsigned* __restrict__ Weu_p,
                              unsigned short* __restrict__ e_bf,
                              float* __restrict__ agg_e, int lastBlock) {
  __shared__ int srow[16], scol[16];
  __shared__ unsigned short Z[16 * 384];
  int e0 = blockIdx.x * 16, tid = threadIdx.x;
  if (tid < 16) { srow[tid] = ei[e0 + tid]; scol[tid] = ei[MM + e0 + tid]; }
  __syncthreads();
  for (int i = tid; i < 16 * 384; i += 256) {
    int m = i / 384, c = i % 384;
    unsigned short v;
    if (c < 128)      v = h_bf[(long long)srow[m] * 128 + c];
    else if (c < 256) v = h_bf[(long long)scol[m] * 128 + (c - 128)];
    else              v = e_bf[(long long)(e0 + m) * 128 + (c - 256)];
    Z[i] = v;
  }
  __syncthreads();
  int wave = tid >> 5, lane = tid & 31, n0 = wave * 16;
  v8f acc = {};
#pragma unroll
  for (int kc = 0; kc < 12; ++kc)
    acc = wmma_bf16(frag_a(Z, 384, kc * 32, lane), frag_bp(Weu_p, kc, wave, 8, lane), acc);
  int n = n0 + (lane & 15), half = (lane >> 4) & 1;
#pragma unroll
  for (int r = 0; r < 8; ++r) {
    int m = r + half * 8;
    long long eidx = (long long)(e0 + m) * 128 + n;
    float v = bf2f(e_bf[eidx]) + siluf(acc[r]);
    e_bf[eidx] = f2bf(v);
    if (lastBlock) atomicAdd(&agg_e[(long long)scol[m] * 128 + n], v);
  }
}

// ---------- final MLP: out = silu(z@W1+b1)@W2+b2, z=[h|agg_e/deg|u[batch]] ----------
__global__ void k_final(const unsigned short* __restrict__ h_bf, const float* __restrict__ agg_e,
                        const float* __restrict__ deg, const unsigned short* __restrict__ u_bf,
                        const int* __restrict__ nbatch,
                        const unsigned* __restrict__ W1_p, const float* __restrict__ b1,
                        const unsigned* __restrict__ W2_p, const float* __restrict__ b2,
                        float* __restrict__ out) {
  __shared__ unsigned short Z[16 * 512];
  __shared__ unsigned short Hid[16 * 128];
  int r0 = blockIdx.x * 16, tid = threadIdx.x;
  for (int i = tid; i < 16 * 512; i += 256) {
    int m = i >> 9, c = i & 511;
    int row = r0 + m;
    unsigned short v;
    if (c < 128) v = h_bf[(long long)row * 128 + c];
    else if (c < 256) {
      float d = deg[row];
      v = f2bf(agg_e[(long long)row * 128 + (c - 128)] * fastrcp(d < 1.f ? 1.f : d));
    } else v = u_bf[(long long)nbatch[row] * 256 + (c - 256)];
    Z[i] = v;
  }
  __syncthreads();
  int wave = tid >> 5, lane = tid & 31, n0 = wave * 16;
  {
    v8f acc = {};
#pragma unroll
    for (int kc = 0; kc < 16; ++kc)
      acc = wmma_bf16(frag_a(Z, 512, kc * 32, lane), frag_bp(W1_p, kc, wave, 8, lane), acc);
    int n = n0 + (lane & 15), half = (lane >> 4) & 1;
#pragma unroll
    for (int r = 0; r < 8; ++r)
      Hid[(r + half * 8) * 128 + n] = f2bf(siluf(acc[r] + b1[n]));
  }
  __syncthreads();
  if (wave == 0) {                      // uniform per wave: EXEC all ones inside
    v8f acc = {};
#pragma unroll
    for (int kc = 0; kc < 4; ++kc)
      acc = wmma_bf16(frag_a(Hid, 128, kc * 32, lane), frag_bp(W2_p, kc, 0, 1, lane), acc);
    int n = lane & 15, half = (lane >> 4) & 1;
    if (n < 3) {
#pragma unroll
      for (int r = 0; r < 8; ++r)
        out[(long long)(r0 + r + half * 8) * 3 + n] = acc[r] + b2[n];
    }
  }
}

// ---------------- host ----------------
extern "C" void kernel_launch(void* const* d_in, const int* in_sizes, int n_in,
                              void* d_out, int out_size, void* d_ws, size_t ws_size,
                              hipStream_t stream) {
  const float* nodes  = (const float*)d_in[0];
  const float* edges  = (const float*)d_in[1];
  const float* virt   = (const float*)d_in[2];
  const int*   ei     = (const int*)d_in[3];
  const int*   nbatch = (const int*)d_in[4];
  /* d_in[5] lengths: unused by reference math */
  const float* t      = (const float*)d_in[6];
  const float* c7     = (const float*)d_in[7];
  const float* x      = (const float*)d_in[8];
  const float* W_nin  = (const float*)d_in[9];
  const float* W_ein  = (const float*)d_in[10];
  const float* W_t    = (const float*)d_in[11];
  const float* Wq     = (const float*)d_in[12];
  const float* Wk     = (const float*)d_in[13];
  const float* Wv     = (const float*)d_in[14];
  const float* Wpos   = (const float*)d_in[15];
  const float* bpos   = (const float*)d_in[16];
  const float* Wg     = (const float*)d_in[17];
  const float* Wo     = (const float*)d_in[18];
  const float* Wvirt  = (const float*)d_in[19];
  const float* Weu    = (const float*)d_in[20];
  const float* W1     = (const float*)d_in[21];
  const float* b1     = (const float*)d_in[22];
  const float* W2     = (const float*)d_in[23];
  const float* b2     = (const float*)d_in[24];
  float* out = (float*)d_out;

  char* p = (char*)d_ws;
  auto alloc = [&](size_t bytes) { char* r = p; p += (bytes + 255) & ~(size_t)255; return r; };

  float*          h        = (float*)alloc((size_t)NN * 128 * 4);
  unsigned short* h_bf     = (unsigned short*)alloc((size_t)NN * 128 * 2);
  unsigned short* e_bf     = (unsigned short*)alloc((size_t)MM * 128 * 2);
  unsigned short* q_bf     = (unsigned short*)alloc((size_t)NN * 128 * 2);
  unsigned short* k_bf     = (unsigned short*)alloc((size_t)NN * 128 * 2);
  unsigned short* v_bf     = (unsigned short*)alloc((size_t)NN * 128 * 2);
  unsigned short* logits_bf= (unsigned short*)alloc((size_t)MM * 128 * 2);
  unsigned short* val_bf   = (unsigned short*)alloc((size_t)MM * 128 * 2);
  unsigned*       nodemax  = (unsigned*)alloc((size_t)NN * 128 * 4);
  float*          expsum   = (float*)alloc((size_t)NN * 128 * 4);
  float*          aggv     = (float*)alloc((size_t)NN * 128 * 4);
  float*          agg_e    = (float*)alloc((size_t)NN * 128 * 4);
  float*          deg      = (float*)alloc((size_t)NN * 4);
  float*          virt_out = (float*)alloc((size_t)GG * 128 * 4);
  unsigned short* u_bf     = (unsigned short*)alloc((size_t)GG * 256 * 2);
  unsigned short* nodes_bf = (unsigned short*)alloc((size_t)NN * 64 * 2);
  unsigned short* edges_bf = (unsigned short*)alloc((size_t)MM * 32 * 2);
  unsigned short* virt_bf  = (unsigned short*)alloc((size_t)GG * 128 * 2);
  float*          tembed   = (float*)alloc(128 * 4);
  // packed (fragment-order) weights, dword units
  unsigned* Wnin_p = (unsigned*)alloc((size_t)2 * 8 * 256 * 4);        // K=64
  unsigned* Wein_p = (unsigned*)alloc((size_t)1 * 8 * 256 * 4);        // K=32(pad)
  unsigned* Wq_p   = (unsigned*)alloc((size_t)2 * 4 * 8 * 256 * 4);    // 2 blocks, K=128
  unsigned* Wk_p   = (unsigned*)alloc((size_t)2 * 4 * 8 * 256 * 4);
  unsigned* Wv_p   = (unsigned*)alloc((size_t)2 * 4 * 8 * 256 * 4);
  unsigned* Wg_p   = (unsigned*)alloc((size_t)2 * 4 * 8 * 256 * 4);
  unsigned* Wo_p   = (unsigned*)alloc((size_t)2 * 4 * 8 * 256 * 4);
  unsigned* Wvi_p  = (unsigned*)alloc((size_t)2 * 4 * 8 * 256 * 4);
  unsigned* Weu_p  = (unsigned*)alloc((size_t)2 * 12 * 8 * 256 * 4);   // K=384
  unsigned* W1_p   = (unsigned*)alloc((size_t)16 * 8 * 256 * 4);       // K=512
  unsigned* W2_p   = (unsigned*)alloc((size_t)4 * 1 * 256 * 4);        // K=128, 1 tile

  auto cvt = [&](const float* s, unsigned short* d, int ri, int ci, int ro, int co) {
    long long tot = (long long)ro * co;
    k_cvt_pad<<<(unsigned)((tot + 255) / 256), 256, 0, stream>>>(s, d, ri, ci, ro, co);
  };
  auto packB = [&](const float* s, unsigned* d, int K, int Cin, int kchunks, int ctiles) {
    long long tot = (long long)kchunks * ctiles * 256;
    k_pack_b<<<(unsigned)((tot + 255) / 256), 256, 0, stream>>>(s, d, K, Cin, kchunks, ctiles);
  };

  // ---- one-time conversions / weight packing ----
  cvt(nodes, nodes_bf, NN, 64, NN, 64);
  cvt(edges, edges_bf, MM, 16, MM, 32);        // K padded to 32
  cvt(virt,  virt_bf,  GG, 128, GG, 128);
  packB(W_nin, Wnin_p, 64, 128, 2, 8);
  packB(W_ein, Wein_p, 16, 128, 1, 8);         // rows 16..31 zero-padded
  for (int b = 0; b < 2; ++b) {
    packB(Wq    + b * 16384, Wq_p  + b * 8192,  128, 128, 4, 8);
    packB(Wk    + b * 16384, Wk_p  + b * 8192,  128, 128, 4, 8);
    packB(Wv    + b * 16384, Wv_p  + b * 8192,  128, 128, 4, 8);
    packB(Wg    + b * 16384, Wg_p  + b * 8192,  128, 128, 4, 8);
    packB(Wo    + b * 16384, Wo_p  + b * 8192,  128, 128, 4, 8);
    packB(Wvirt + b * 16384, Wvi_p + b * 8192,  128, 128, 4, 8);
    packB(Weu   + b * 49152, Weu_p + b * 24576, 384, 128, 12, 8);
  }
  packB(W1, W1_p, 512, 128, 16, 8);
  packB(W2, W2_p, 128, 3, 4, 1);               // cols 3..15 zero-padded

  k_tembed<<<1, 128, 0, stream>>>(t, W_t, tembed);
  k_upack<<<GG, 256, 0, stream>>>(virt, c7, t, u_bf);

  // ---- embeddings ----
  k_embed_nodes<<<NN / 16, 256, 0, stream>>>(nodes_bf, Wnin_p, tembed, h, h_bf);
  k_embed_edges<<<MM / 16, 256, 0, stream>>>(edges_bf, Wein_p, e_bf);

  hipMemsetAsync(agg_e, 0, (size_t)NN * 128 * 4, stream);
  hipMemsetAsync(deg, 0, (size_t)NN * 4, stream);
  k_deg<<<(MM + 255) / 256, 256, 0, stream>>>(ei, deg);

  // ---- two point-transformer blocks ----
  for (int b = 0; b < 2; ++b) {
    hipMemsetAsync(nodemax, 0, (size_t)NN * 128 * 4, stream);  // key 0 == -NaN sentinel
    hipMemsetAsync(expsum, 0, (size_t)NN * 128 * 4, stream);
    hipMemsetAsync(aggv, 0, (size_t)NN * 128 * 4, stream);

    k_qkv<<<NN / 16, 256, 0, stream>>>(h_bf, Wq_p + b * 8192, Wk_p + b * 8192,
                                       Wv_p + b * 8192, q_bf, k_bf, v_bf);
    k_virt<<<GG / 16, 256, 0, stream>>>(virt_bf, Wvi_p + b * 8192, virt_out);
    k_edge_pass1<<<MM / 16, 256, 0, stream>>>(ei, x, q_bf, k_bf, v_bf, e_bf,
                                              Wpos + b * 384, bpos + b * 128,
                                              Wg_p + b * 8192, logits_bf, val_bf, nodemax);
    k_edge_pass2<<<4096, 256, 0, stream>>>(ei, logits_bf, val_bf, nodemax, expsum, aggv);
    k_node_update<<<NN / 16, 256, 0, stream>>>(aggv, expsum, Wo_p + b * 8192,
                                               virt_out, nbatch, h, h_bf);
    k_edge_update<<<MM / 16, 256, 0, stream>>>(ei, h_bf, Weu_p + b * 24576,
                                               e_bf, agg_e, (b == 1) ? 1 : 0);
  }

  // ---- final node MLP ----
  k_final<<<NN / 16, 256, 0, stream>>>(h_bf, agg_e, deg, u_bf, nbatch,
                                       W1_p, b1, W2_p, b2, out);
}
